// MultiheadAttention_3942779977850
// MI455X (gfx1250) — matrix operand order repaired and checked by reading, not compiled, not run on hardware
//
#include <hip/hip_runtime.h>
#include <hip/hip_bf16.h>

// ---------------------------------------------------------------------------
// Problem constants (match reference)
// ---------------------------------------------------------------------------
#define BB   8
#define CC   512
#define HH   8            // heads
#define DD   64           // head dim
#define NN   1024         // tokens = H*W
#define OO   1536         // 3*C
#define CNT  (CC * NN)    // elements per batch for groupnorm (C*H*W)
#define EPSV 1e-5f
#define SCALE 0.044194173824159216f   // 512^-0.5

typedef __attribute__((ext_vector_type(16))) _Float16 v16h;
typedef __attribute__((ext_vector_type(8)))  _Float16 v8h;
typedef __attribute__((ext_vector_type(8)))  float    v8f;
typedef __attribute__((ext_vector_type(4)))  unsigned int u32x4;
typedef __attribute__((ext_vector_type(8)))  int      i32x8;
typedef __attribute__((ext_vector_type(4)))  int      i32x4;

static __device__ inline v8f vzero() {
    v8f z = {0.f, 0.f, 0.f, 0.f, 0.f, 0.f, 0.f, 0.f};
    return z;
}

// A-fragment (16x32, f16): lane holds row = lane&15; two contiguous 8-half runs
// at K-offset (lane>>4)*8 and (lane>>4)*8 + 16.   ld = row stride in halves.
static __device__ inline v16h load_a_frag(const _Float16* base, int ld) {
    int lane = threadIdx.x & 31;
    int row  = lane & 15;
    int ko   = (lane >> 4) * 8;
    const _Float16* p = base + (size_t)row * ld + ko;
    v16h r;
    *((v8h*)&r)     = *(const v8h*)(p);
    *((v8h*)&r + 1) = *(const v8h*)(p + 16);
    return r;
}

// B-fragment (32x16, f16): lane holds column = lane&15; two contiguous 8-half runs
// at K-offset (lane>>4)*8 and (lane>>4)*8 + 16.  ld = column stride in halves.
static __device__ inline v16h load_b_frag(const _Float16* base, int ld) {
    int lane = threadIdx.x & 31;
    int col  = lane & 15;
    int ko   = (lane >> 4) * 8;
    const _Float16* p = base + (size_t)col * ld + ko;
    v16h r;
    *((v8h*)&r)     = *(const v8h*)(p);
    *((v8h*)&r + 1) = *(const v8h*)(p + 16);
    return r;
}

static __device__ inline v8f wmma32(v16h a, v16h b, v8f c) {
    return __builtin_amdgcn_wmma_f32_16x16x32_f16(
        /*neg_a=*/false, a, /*neg_b=*/false, b,
        /*c_mod=*/(short)0, c, /*reuse_a=*/false, /*reuse_b=*/false);
}

// ---------------------------------------------------------------------------
// Tensor Data Mover: DMA a 2-D tile (f16 elements) from global to LDS.
// D# built per CDNA5 ISA 8.3/8.4: group0 = count/lds_addr/global_addr/type,
// group1 = data_size + tensor dims/strides + tile dims. Groups 2/3 = NULL
// (2-D tensor). All dims/strides in 2-byte element units.
// ---------------------------------------------------------------------------
static __device__ inline void tdm_load_2d(unsigned lds_off, const void* gptr,
                                          unsigned tile_d0, unsigned tile_d1,
                                          unsigned tensor_d0, unsigned tensor_d1,
                                          unsigned stride0, unsigned stride1) {
    unsigned long long ga = (unsigned long long)(uintptr_t)gptr;
    u32x4 g0;
    g0[0] = 1u;                                        // count=1 (valid user D#)
    g0[1] = lds_off;                                   // lds_addr (bytes)
    g0[2] = (unsigned)(ga & 0xffffffffu);              // global_addr[31:0]
    g0[3] = (unsigned)((ga >> 32) & 0x01ffffffu)       // global_addr[56:32]
            | (2u << 30);                              // type = 2 ("image")
    i32x8 g1;
    g1[0] = (int)(1u << 16);                           // data_size = 1 (2 bytes)
    g1[1] = (int)((tensor_d0 & 0xffffu) << 16);        // tensor_dim0[15:0]
    g1[2] = (int)(((tensor_d0 >> 16) & 0xffffu)        // tensor_dim0[31:16]
                  | ((tensor_d1 & 0xffffu) << 16));    // tensor_dim1[15:0]
    g1[3] = (int)(((tensor_d1 >> 16) & 0xffffu)        // tensor_dim1[31:16]
                  | ((tile_d0 & 0xffffu) << 16));      // tile_dim0
    g1[4] = (int)(tile_d1 & 0xffffu);                  // tile_dim1 (tile_dim2=0)
    g1[5] = (int)stride0;                              // tensor_dim0_stride[31:0]
    g1[6] = (int)((stride1 & 0xffffu) << 16);          // stride0[47:32]=0 | stride1[15:0]
    g1[7] = (int)(stride1 >> 16);                      // stride1[47:16]
    i32x4 gz = {0, 0, 0, 0};
#if defined(__clang_major__) && (__clang_major__ >= 23)
    i32x8 gz8 = {0, 0, 0, 0, 0, 0, 0, 0};
    __builtin_amdgcn_tensor_load_to_lds(g0, g1, gz, gz, gz8, 0);
#else
    __builtin_amdgcn_tensor_load_to_lds(g0, g1, gz, gz, 0);
#endif
}

// ---------------------------------------------------------------------------
// Kernel 1: per-batch mean / rsqrt(var) over C*H*W
// ---------------------------------------------------------------------------
__global__ __launch_bounds__(256) void gn_stats_kernel(const float* __restrict__ x,
                                                       float* __restrict__ stats) {
    int b = blockIdx.x;
    const float* p = x + (size_t)b * CNT;
    float s = 0.f, s2 = 0.f;
    for (int i = threadIdx.x; i < CNT; i += 256) {
        float v = p[i];
        s  += v;
        s2 += v * v;
    }
    __shared__ float ls[256], lq[256];
    ls[threadIdx.x] = s;
    lq[threadIdx.x] = s2;
    __syncthreads();
    for (int off = 128; off > 0; off >>= 1) {
        if (threadIdx.x < off) {
            ls[threadIdx.x] += ls[threadIdx.x + off];
            lq[threadIdx.x] += lq[threadIdx.x + off];
        }
        __syncthreads();
    }
    if (threadIdx.x == 0) {
        float mu  = ls[0] / (float)CNT;
        float var = lq[0] / (float)CNT - mu * mu;
        stats[2 * b]     = mu;
        stats[2 * b + 1] = rsqrtf(var + EPSV);
    }
}

// ---------------------------------------------------------------------------
// Kernel 2: normalize + affine + cast to f16, store transposed xnT[b][n][c]
// ---------------------------------------------------------------------------
__global__ __launch_bounds__(256) void gn_norm_kernel(const float* __restrict__ x,
                                                      const float* __restrict__ gamma,
                                                      const float* __restrict__ beta,
                                                      const float* __restrict__ stats,
                                                      _Float16* __restrict__ xnT) {
    size_t idx = (size_t)blockIdx.x * 256 + threadIdx.x;
    if (idx >= (size_t)BB * CNT) return;
    int b   = (int)(idx >> 19);        // C*N = 2^19
    int rem = (int)(idx & (CNT - 1));
    int c   = rem >> 10;               // N = 2^10
    int n   = rem & (NN - 1);
    float mu = stats[2 * b], rs = stats[2 * b + 1];
    float v  = (x[idx] - mu) * rs * gamma[c] + beta[c];
    xnT[(size_t)b * CNT + (size_t)n * CC + c] = (_Float16)v;
}

// ---------------------------------------------------------------------------
// Kernel 3: f32 -> f16 weight cast
// ---------------------------------------------------------------------------
__global__ __launch_bounds__(256) void cast_f16_kernel(const float* __restrict__ src,
                                                       _Float16* __restrict__ dst, int n) {
    int i = blockIdx.x * 256 + threadIdx.x;
    if (i < n) dst[i] = (_Float16)src[i];
}

// ---------------------------------------------------------------------------
// Kernel 4: QKV GEMM with TDM-staged B tiles.
//   qkv[o][n] = sum_c W[o][c]*xnT[n][c] + bias[o]
// Block tile 128(M) x 128(N); 8 waves, each wave owns 16 M-rows x 128 N-cols.
// B tile (128 tokens x 32 halves) is DMA'd to LDS once per block per k-chunk
// (double buffered; DMA of chunk k+1 overlaps WMMAs of chunk k). All 8 B
// fragments are hoisted into registers before the WMMA chain so the ds_loads
// clause together instead of serializing each WMMA behind s_wait_dscnt 0.
// Output scattered into attention-friendly layouts:
//   Q,K : [b*H+h][n][d]   (d contiguous)
//   V   : [b*H+h][d][n]   (n contiguous)
// ---------------------------------------------------------------------------
__global__ __launch_bounds__(256) void qkv_gemm_kernel(const _Float16* __restrict__ W,
                                                       const float* __restrict__ bias,
                                                       const _Float16* __restrict__ xnT,
                                                       _Float16* __restrict__ qbuf,
                                                       _Float16* __restrict__ kbuf,
                                                       _Float16* __restrict__ vbuf) {
    int b     = blockIdx.z;
    int mtile = blockIdx.y * 128;
    int ntile = blockIdx.x * 128;
    int wave  = threadIdx.x >> 5;
    int lane  = threadIdx.x & 31;
    int mw    = mtile + wave * 16;

    const _Float16* Arow  = W + (size_t)mw * CC;
    const _Float16* Bbase = xnT + (size_t)b * CNT + (size_t)ntile * CC;

    __shared__ __align__(16) _Float16 btile[2][128 * 32];   // 2 x 8 KB

    v8f acc[8];
#pragma unroll
    for (int j = 0; j < 8; ++j) acc[j] = vzero();

    // prime chunk 0
    if (wave == 0)
        tdm_load_2d((unsigned)(uintptr_t)&btile[0][0], Bbase,
                    /*tile*/ 32, 128, /*tensor*/ CC, NN, /*strides*/ CC, CC);

    for (int kc = 0; kc < CC / 32; ++kc) {
        int k = kc * 32;
        if (wave == 0) {
            if (kc + 1 < CC / 32) {
                tdm_load_2d((unsigned)(uintptr_t)&btile[(kc + 1) & 1][0],
                            Bbase + k + 32, 32, 128, CC, NN, CC, CC);
                __builtin_amdgcn_s_wait_tensorcnt(1);   // oldest (chunk kc) done
            } else {
                __builtin_amdgcn_s_wait_tensorcnt(0);
            }
        }
        __syncthreads();

        if (k + 32 < CC) __builtin_prefetch(Arow + k + 32, 0, 1);
        v16h a = load_a_frag(Arow + k, CC);
        const _Float16* bl = &btile[kc & 1][0];
        // hoist all B fragments: lets ds_loads clause, then WMMAs run back-to-back
        v16h bf[8];
#pragma unroll
        for (int j = 0; j < 8; ++j)
            bf[j] = load_b_frag(bl + (size_t)(j * 16) * 32, 32);
#pragma unroll
        for (int j = 0; j < 8; ++j)
            acc[j] = wmma32(a, bf[j], acc[j]);
        __syncthreads();   // all waves done with this buffer before its reuse
    }

    // epilogue: bias add + scatter
    int mrow0 = mw + ((lane >> 4) * 8);     // 8 consecutive output rows
    float br[8];
#pragma unroll
    for (int r = 0; r < 8; ++r) br[r] = bias[mrow0 + r];

    int region = mw / CC;                   // 0=Q 1=K 2=V (uniform per block)
    int olocal = mw % CC;
    int hh     = olocal / DD;
    int d0     = (olocal % DD) + ((lane >> 4) * 8);
    size_t bh  = (size_t)(b * HH + hh);

#pragma unroll
    for (int j = 0; j < 8; ++j) {
        int tok = ntile + j * 16 + (lane & 15);
        if (region == 2) {                  // V : [bh][d][n], strided scatter
#pragma unroll
            for (int r = 0; r < 8; ++r)
                vbuf[(bh * DD + d0 + r) * NN + tok] = (_Float16)(acc[j][r] + br[r]);
        } else {                            // Q/K : [bh][n][d], contiguous v8h
            v8h hv;
#pragma unroll
            for (int r = 0; r < 8; ++r) hv[r] = (_Float16)(acc[j][r] + br[r]);
            _Float16* dst = (region == 0 ? qbuf : kbuf) + (bh * NN + tok) * DD + d0;
            *(v8h*)dst = hv;
        }
    }
}

// ---------------------------------------------------------------------------
// Kernel 5: flash attention.  grid = (N/128, B*H), 256 threads (8 waves).
// Each wave: 16 query rows, streams key/value tiles of 32, online softmax.
// ---------------------------------------------------------------------------
__global__ __launch_bounds__(256) void attn_kernel(const _Float16* __restrict__ qbuf,
                                                   const _Float16* __restrict__ kbuf,
                                                   const _Float16* __restrict__ vbuf,
                                                   _Float16* __restrict__ valsT) {
    int bh   = blockIdx.y;
    int b    = bh / HH, h = bh % HH;
    int wave = threadIdx.x >> 5;
    int lane = threadIdx.x & 31;
    int qrow = blockIdx.x * 128 + wave * 16;

    const _Float16* Qb = qbuf + ((size_t)bh * NN + qrow) * DD;
    const _Float16* Kb = kbuf + (size_t)bh * NN * DD;
    const _Float16* Vb = vbuf + (size_t)bh * DD * NN;

    // Q fragments (fixed for entire kv loop): K-dim = D = 64 -> two chunks
    v16h q0 = load_a_frag(Qb + 0, DD);
    v16h q1 = load_a_frag(Qb + 32, DD);

    v8f o[4];
#pragma unroll
    for (int dt = 0; dt < 4; ++dt) o[dt] = vzero();
    float mi[8], li[8];
#pragma unroll
    for (int r = 0; r < 8; ++r) { mi[r] = -3.0e38f; li[r] = 0.f; }

    __shared__ __align__(16) _Float16 lds_p[8][16 * 32];  // per-wave P staging
    _Float16* pw = &lds_p[wave][0];

    for (int kt = 0; kt < NN; kt += 32) {
        // ---- S = (Q^T K) * scale : two 16x16 tiles, K-dim 64 ----
        v16h kb00 = load_b_frag(Kb + (size_t)kt * DD + 0, DD);
        v16h kb01 = load_b_frag(Kb + (size_t)kt * DD + 32, DD);
        v16h kb10 = load_b_frag(Kb + (size_t)(kt + 16) * DD + 0, DD);
        v16h kb11 = load_b_frag(Kb + (size_t)(kt + 16) * DD + 32, DD);
        // prefetch V fragments for the PV stage of this key tile
        v16h vf[4];
#pragma unroll
        for (int dt = 0; dt < 4; ++dt)
            vf[dt] = load_b_frag(Vb + (size_t)(dt * 16) * NN + kt, NN);

        v8f s0 = vzero(), s1 = vzero();
        s0 = wmma32(q0, kb00, s0);
        s0 = wmma32(q1, kb01, s0);
        s1 = wmma32(q0, kb10, s1);
        s1 = wmma32(q1, kb11, s1);

        // ---- online softmax: row max over 16 lanes of each half-group ----
        float mnew[8], corr[8];
#pragma unroll
        for (int r = 0; r < 8; ++r) {
            float a0 = s0[r] * SCALE, a1 = s1[r] * SCALE;
            s0[r] = a0; s1[r] = a1;
            float t = fmaxf(a0, a1);
            t = fmaxf(t, __shfl_xor(t, 1, 32));
            t = fmaxf(t, __shfl_xor(t, 2, 32));
            t = fmaxf(t, __shfl_xor(t, 4, 32));
            t = fmaxf(t, __shfl_xor(t, 8, 32));
            mnew[r] = fmaxf(mi[r], t);
            corr[r] = __expf(mi[r] - mnew[r]);
        }
#pragma unroll
        for (int r = 0; r < 8; ++r) {
            float p0 = __expf(s0[r] - mnew[r]);
            float p1 = __expf(s1[r] - mnew[r]);
            float rs = p0 + p1;
            rs += __shfl_xor(rs, 1, 32);
            rs += __shfl_xor(rs, 2, 32);
            rs += __shfl_xor(rs, 4, 32);
            rs += __shfl_xor(rs, 8, 32);
            li[r] = li[r] * corr[r] + rs;
            mi[r] = mnew[r];
            // stage P (C-frag layout -> row-major LDS tile 16x32)
            int row = ((lane >> 4) * 8) + r;
            int col = lane & 15;
            pw[row * 32 + col]      = (_Float16)p0;
            pw[row * 32 + 16 + col] = (_Float16)p1;
            // rescale running output
            o[0][r] *= corr[r];
            o[1][r] *= corr[r];
            o[2][r] *= corr[r];
            o[3][r] *= corr[r];
        }
        // re-read P as an A-fragment (same-wave LDS RAW is in-order)
        v16h pf = load_a_frag(pw, 32);
        // ---- O += P * V  (4 tiles along d) ----
#pragma unroll
        for (int dt = 0; dt < 4; ++dt)
            o[dt] = wmma32(pf, vf[dt], o[dt]);
    }

    // ---- epilogue: divide by row sums, scatter into valsT[b][n][h*64+d] ----
    float inv[8];
#pragma unroll
    for (int r = 0; r < 8; ++r) inv[r] = 1.f / li[r];
    int mb = (lane >> 4) * 8;
#pragma unroll
    for (int dt = 0; dt < 4; ++dt) {
        int d = dt * 16 + (lane & 15);
#pragma unroll
        for (int r = 0; r < 8; ++r) {
            int tok = qrow + mb + r;
            valsT[((size_t)b * NN + tok) * CC + h * DD + d] =
                (_Float16)(o[dt][r] * inv[r]);
        }
    }
}

// ---------------------------------------------------------------------------
// Kernel 6: output projection with TDM-staged B tiles.
//   out[b][o][n] = sum_c Wp[o][c]*valsT[n][c] + bp[o]
// ---------------------------------------------------------------------------
__global__ __launch_bounds__(256) void proj_gemm_kernel(const _Float16* __restrict__ W,
                                                        const float* __restrict__ bias,
                                                        const _Float16* __restrict__ valsT,
                                                        float* __restrict__ out) {
    int b     = blockIdx.z;
    int mtile = blockIdx.y * 128;
    int ntile = blockIdx.x * 128;
    int wave  = threadIdx.x >> 5;
    int lane  = threadIdx.x & 31;
    int mw    = mtile + wave * 16;

    const _Float16* Arow  = W + (size_t)mw * CC;
    const _Float16* Bbase = valsT + (size_t)b * CNT + (size_t)ntile * CC;

    __shared__ __align__(16) _Float16 btile[2][128 * 32];   // 2 x 8 KB

    v8f acc[8];
#pragma unroll
    for (int j = 0; j < 8; ++j) acc[j] = vzero();

    if (wave == 0)
        tdm_load_2d((unsigned)(uintptr_t)&btile[0][0], Bbase,
                    32, 128, CC, NN, CC, CC);

    for (int kc = 0; kc < CC / 32; ++kc) {
        int k = kc * 32;
        if (wave == 0) {
            if (kc + 1 < CC / 32) {
                tdm_load_2d((unsigned)(uintptr_t)&btile[(kc + 1) & 1][0],
                            Bbase + k + 32, 32, 128, CC, NN, CC, CC);
                __builtin_amdgcn_s_wait_tensorcnt(1);
            } else {
                __builtin_amdgcn_s_wait_tensorcnt(0);
            }
        }
        __syncthreads();

        if (k + 32 < CC) __builtin_prefetch(Arow + k + 32, 0, 1);
        v16h a = load_a_frag(Arow + k, CC);
        const _Float16* bl = &btile[kc & 1][0];
        v16h bf[8];
#pragma unroll
        for (int j = 0; j < 8; ++j)
            bf[j] = load_b_frag(bl + (size_t)(j * 16) * 32, 32);
#pragma unroll
        for (int j = 0; j < 8; ++j)
            acc[j] = wmma32(a, bf[j], acc[j]);
        __syncthreads();
    }

    int mrow0 = mw + ((lane >> 4) * 8);
    float br[8];
#pragma unroll
    for (int r = 0; r < 8; ++r) br[r] = bias[mrow0 + r];

#pragma unroll
    for (int j = 0; j < 8; ++j) {
        int tok = ntile + j * 16 + (lane & 15);
#pragma unroll
        for (int r = 0; r < 8; ++r)
            out[((size_t)b * CC + mrow0 + r) * NN + tok] = acc[j][r] + br[r];
    }
}

// ---------------------------------------------------------------------------
// Host-side launch
// ---------------------------------------------------------------------------
extern "C" void kernel_launch(void* const* d_in, const int* in_sizes, int n_in,
                              void* d_out, int out_size, void* d_ws, size_t ws_size,
                              hipStream_t stream) {
    const float* x      = (const float*)d_in[0];
    const float* gamma  = (const float*)d_in[1];
    const float* beta   = (const float*)d_in[2];
    const float* w_qkv  = (const float*)d_in[3];
    const float* b_qkv  = (const float*)d_in[4];
    const float* w_proj = (const float*)d_in[5];
    const float* b_proj = (const float*)d_in[6];
    float* out = (float*)d_out;

    // workspace layout (bytes, 256-aligned)
    char* ws = (char*)d_ws;
    size_t off = 0;
    float*    stats  = (float*)(ws + off);     off += 256;
    _Float16* wq16   = (_Float16*)(ws + off);  off += (size_t)OO * CC * 2;          // 1.5 MB
    _Float16* wp16   = (_Float16*)(ws + off);  off += (size_t)CC * CC * 2;          // 0.5 MB
    _Float16* xnT    = (_Float16*)(ws + off);  off += (size_t)BB * CNT * 2;         // 8 MB
    _Float16* qbuf   = (_Float16*)(ws + off);  off += (size_t)BB * HH * NN * DD * 2;// 8 MB
    _Float16* kbuf   = (_Float16*)(ws + off);  off += (size_t)BB * HH * NN * DD * 2;// 8 MB
    _Float16* vbuf   = (_Float16*)(ws + off);  off += (size_t)BB * HH * NN * DD * 2;// 8 MB
    _Float16* valsT  = (_Float16*)(ws + off);  off += (size_t)BB * CNT * 2;         // 8 MB

    // 1) groupnorm statistics
    gn_stats_kernel<<<BB, 256, 0, stream>>>(x, stats);

    // 2) normalize + transpose + cast
    {
        size_t total  = (size_t)BB * CNT;
        int    blocks = (int)((total + 255) / 256);
        gn_norm_kernel<<<blocks, 256, 0, stream>>>(x, gamma, beta, stats, xnT);
    }

    // 3) weight casts
    cast_f16_kernel<<<(OO * CC + 255) / 256, 256, 0, stream>>>(w_qkv, wq16, OO * CC);
    cast_f16_kernel<<<(CC * CC + 255) / 256, 256, 0, stream>>>(w_proj, wp16, CC * CC);

    // 4) QKV projection GEMM (WMMA + TDM)
    qkv_gemm_kernel<<<dim3(NN / 128, OO / 128, BB), 256, 0, stream>>>(
        wq16, b_qkv, xnT, qbuf, kbuf, vbuf);

    // 5) flash attention (WMMA)
    attn_kernel<<<dim3(NN / 128, BB * HH), 256, 0, stream>>>(qbuf, kbuf, vbuf, valsT);

    // 6) output projection GEMM (WMMA + TDM)
    proj_gemm_kernel<<<dim3(NN / 128, CC / 128, BB), 256, 0, stream>>>(
        wp16, b_proj, valsT, out);
}
